// ARM_57226144252650
// MI455X (gfx1250) — matrix operand order repaired and checked by reading, not compiled, not run on hardware
//
#include <hip/hip_runtime.h>
#include <hip/hip_bf16.h>

// ---------------------------------------------------------------------------
// Vision-Mamba forward for MI455X (gfx1250, wave32).
//  * GEMMs: v_wmma_f32_16x16x32_bf16, fp32 accumulate, packed native
//    fp32->bf16 converts (v_cvt_pk_bf16_f32 via inline asm) during LDS
//    staging, ping-pong double-buffered K loop (next-tile global loads issue
//    before the WMMAs, LDS stores after).
//  * Patch embed: GLOBAL_LOAD_ASYNC_TO_LDS_B32 (CDNA5 async copy, ASYNCcnt).
//  * Weights (~160MB fp32) are L2-resident (192MB), so steady state is
//    matrix-core bound; serial scan parallelized over 3072 (b,d) channels.
// ---------------------------------------------------------------------------

#define BATCH 8
#define SEQ   196            // (224/16)^2
#define DIM   384
#define DEPTH 24
#define NST   16
#define KCONV 4
#define RRANK 24
#define HID   1024           // 384*4*2/3
#define TOK   (BATCH * SEQ)  // 1568 (multiple of 16)

typedef __attribute__((ext_vector_type(16))) __bf16   bf16x16;
typedef __attribute__((ext_vector_type(8)))  float    f32x8;
typedef __attribute__((ext_vector_type(4)))  float    f32x4;
typedef __attribute__((ext_vector_type(2)))  unsigned u32x2;
typedef __attribute__((ext_vector_type(4)))  unsigned u32x4;

// Packed fp32->bf16: exactly one v_cvt_pk_bf16_f32 per 32-bit result.
__device__ __forceinline__ unsigned packbf2(float a, float b) {
#if defined(__AMDGCN__)
  unsigned r;
  asm("v_cvt_pk_bf16_f32 %0, %1, %2" : "=v"(r) : "v"(a), "v"(b));
  return r;
#else
  union { __bf16 h[2]; unsigned u; } p;
  p.h[0] = (__bf16)a; p.h[1] = (__bf16)b;
  return p.u;
#endif
}
__device__ __forceinline__ unsigned short f2bf(float f) {
  union { __bf16 h; unsigned short s; } c;
  c.h = (__bf16)f;
  return c.s;
}
__device__ __forceinline__ float silu(float v) { return v / (1.f + __expf(-v)); }

// ---------------------------------------------------------------------------
// WMMA GEMM:  C[M x Nw] (fp32) (+)= A[M x K] * W[Nw x K]^T  (+ bias)
// Block tile 128(M) x 64(N); 8 waves, each a 32x32 tile = 4 WMMA accums.
// A staged row-major bf16 (stride 40 shorts); W staged TRANSPOSED [k][n]
// (stride 72 shorts); each lane gathers its 16 B operand values one bf16 per k (stride 72).
// Double-buffered: global loads for k+32 are in flight while k computes.
// ---------------------------------------------------------------------------
enum { EPI_STORE = 0, EPI_SILU = 1, EPI_RESADD = 2 };

__device__ __forceinline__ void lds_store_a(unsigned short* Abuf, int ar_st,
                                            int ac_st, int j, f32x4 f) {
  u32x2 p;
  p.x = packbf2(f.x, f.y);
  p.y = packbf2(f.z, f.w);
  *(u32x2*)&Abuf[ar_st * 40 + ac_st + 4 * j] = p;
}
__device__ __forceinline__ void lds_store_bT(unsigned short* Bbuf, int bk_st,
                                             int bn_st, int j, f32x4 f) {
  Bbuf[(bk_st + 4 * j + 0) * 72 + bn_st] = f2bf(f.x);
  Bbuf[(bk_st + 4 * j + 1) * 72 + bn_st] = f2bf(f.y);
  Bbuf[(bk_st + 4 * j + 2) * 72 + bn_st] = f2bf(f.z);
  Bbuf[(bk_st + 4 * j + 3) * 72 + bn_st] = f2bf(f.w);
}

template <int EPI>
__global__ __launch_bounds__(256) void gemm_wmma_bf16(
    const float* __restrict__ A, const float* __restrict__ W,
    const float* __restrict__ bias, float* __restrict__ C,
    int M, int Nw, int K, int ldc)
{
  __shared__ __align__(16) unsigned short As[2][128 * 40];
  __shared__ __align__(16) unsigned short Bs[2][32 * 72];

  const int tid  = threadIdx.x;
  const int m0   = blockIdx.x * 128;
  const int n0   = blockIdx.y * 64;
  const int wid  = tid >> 5, lane = tid & 31;
  const int wm   = (wid & 3) * 32;   // wave M offset in block tile
  const int wn   = (wid >> 2) * 32;  // wave N offset in block tile

  f32x8 acc[2][2] = {};

  // A staging assignment: thread -> (row, col16-chunk)
  const int  ar_st = tid >> 1;
  const int  ac_st = (tid & 1) * 16;
  const int  amr   = (m0 + ar_st < M) ? (m0 + ar_st) : (M - 1);
  const long amrow = (long)amr * K;

  // B staging assignment: thread -> (n row, k8-chunk)
  const int  bn_st  = tid >> 2;
  const int  bk_st  = (tid & 3) * 8;
  const bool bvalid = (n0 + bn_st) < Nw;
  const long bnrow  = bvalid ? (long)(n0 + bn_st) * K : 0;

  // ---- prologue: stage k=0 into buffer 0 ----
  {
    const f32x4* srcA = (const f32x4*)(A + amrow + ac_st);
#pragma unroll
    for (int j = 0; j < 4; ++j) lds_store_a(As[0], ar_st, ac_st, j, srcA[j]);
#pragma unroll
    for (int j = 0; j < 2; ++j) {
      f32x4 f = {0.f, 0.f, 0.f, 0.f};
      if (bvalid) f = ((const f32x4*)(W + bnrow + bk_st))[j];
      lds_store_bT(Bs[0], bk_st, bn_st, j, f);
    }
  }
  __syncthreads();

  int buf = 0;
  for (int k0 = 0; k0 < K; k0 += 32, buf ^= 1) {
    const bool next = (k0 + 32) < K;     // uniform across the block

    // (a) issue next-tile global loads (latency hidden behind WMMAs)
    f32x4 an[4], bn2[2];
    if (next) {
      const f32x4* srcA = (const f32x4*)(A + amrow + k0 + 32 + ac_st);
#pragma unroll
      for (int j = 0; j < 4; ++j) an[j] = srcA[j];
#pragma unroll
      for (int j = 0; j < 2; ++j) {
        f32x4 f = {0.f, 0.f, 0.f, 0.f};
        if (bvalid) f = ((const f32x4*)(W + bnrow + k0 + 32 + bk_st))[j];
        bn2[j] = f;
      }
      if (k0 + 64 < K) {                 // pull k+64 toward L2/L0
        __builtin_prefetch(A + amrow + k0 + 64 + ac_st, 0, 1);
        if (bvalid) __builtin_prefetch(W + bnrow + k0 + 64 + bk_st, 0, 1);
      }
    }

    // (b) operands from current buffer + WMMAs
    union ABU { u32x4 q[2]; bf16x16 v; };
    ABU a[2], b[2];
    const int ak = (lane < 16) ? 0 : 8;  // A: half-wave K interleave
#pragma unroll
    for (int mt = 0; mt < 2; ++mt) {
      const int ar = wm + mt * 16 + (lane & 15);
      a[mt].q[0] = *(const u32x4*)&As[buf][ar * 40 + ak];       // K ak..ak+7
      a[mt].q[1] = *(const u32x4*)&As[buf][ar * 40 + ak + 16];  // K ak+16..+23
    }
#pragma unroll
    for (int nt = 0; nt < 2; ++nt) {                            // B: lane&15 = N, K as A
      const int bn = wn + nt * 16 + (lane & 15);
#pragma unroll
      for (int s = 0; s < 16; ++s) {
        const int kk = (s < 8) ? (ak + s) : (16 + ak + s - 8);
        b[nt].v[s] = __builtin_bit_cast(__bf16, Bs[buf][kk * 72 + bn]);
      }
    }
#pragma unroll
    for (int mt = 0; mt < 2; ++mt)
#pragma unroll
      for (int nt = 0; nt < 2; ++nt)
        acc[mt][nt] = __builtin_amdgcn_wmma_f32_16x16x32_bf16(
            false, a[mt].v, false, b[nt].v, (short)0, acc[mt][nt], false, false);

    // (c) convert + store next tile into the other buffer
    if (next) {
#pragma unroll
      for (int j = 0; j < 4; ++j) lds_store_a(As[buf ^ 1], ar_st, ac_st, j, an[j]);
#pragma unroll
      for (int j = 0; j < 2; ++j) lds_store_bT(Bs[buf ^ 1], bk_st, bn_st, j, bn2[j]);
    }
    // (d) one barrier per k-step
    __syncthreads();
  }

  // ---- epilogue: C layout lane&15 = N, VGPR v = M (lanes>=16: M+8) ----
#pragma unroll
  for (int nt = 0; nt < 2; ++nt) {
    const int n = n0 + wn + nt * 16 + (lane & 15);
    if (n >= Nw) continue;
    const float bv = bias ? bias[n] : 0.f;
#pragma unroll
    for (int mt = 0; mt < 2; ++mt) {
      const int mbase = m0 + wm + mt * 16 + ((lane < 16) ? 0 : 8);
#pragma unroll
      for (int v = 0; v < 8; ++v) {
        const int m = mbase + v;
        if (m >= M) continue;
        float val = acc[mt][nt][v] + bv;
        float* dst = C + (long)m * ldc + n;
        if (EPI == EPI_SILU)   val = silu(val);
        if (EPI == EPI_RESADD) val += *dst;
        *dst = val;
      }
    }
  }
}

// ---------------------------------------------------------------------------
// LayerNorm over last dim (384). One block (128 threads) per row.
// ---------------------------------------------------------------------------
__global__ __launch_bounds__(128) void ln_kernel(
    const float* __restrict__ x, const float* __restrict__ w,
    const float* __restrict__ b, float* __restrict__ o)
{
  __shared__ float red[128];
  const int row = blockIdx.x, tid = threadIdx.x;
  const float* xr = x + (long)row * DIM;
  float v0 = xr[tid], v1 = xr[tid + 128], v2 = xr[tid + 256];
  red[tid] = v0 + v1 + v2; __syncthreads();
  for (int off = 64; off > 0; off >>= 1) {
    if (tid < off) red[tid] += red[tid + off];
    __syncthreads();
  }
  const float mean = red[0] * (1.f / DIM);
  __syncthreads();
  const float d0 = v0 - mean, d1 = v1 - mean, d2 = v2 - mean;
  red[tid] = d0 * d0 + d1 * d1 + d2 * d2; __syncthreads();
  for (int off = 64; off > 0; off >>= 1) {
    if (tid < off) red[tid] += red[tid + off];
    __syncthreads();
  }
  const float rstd = rsqrtf(red[0] * (1.f / DIM) + 1e-5f);
  float* orow = o + (long)row * DIM;
  orow[tid]       = d0 * rstd * w[tid]       + b[tid];
  orow[tid + 128] = d1 * rstd * w[tid + 128] + b[tid + 128];
  orow[tid + 256] = d2 * rstd * w[tid + 256] + b[tid + 256];
}

// ---------------------------------------------------------------------------
// Patch embed: one block per token, 384 threads (one per output channel).
// The 768-element patch is DMA'd into LDS with per-lane async copies
// (GLOBAL_LOAD_ASYNC_TO_LDS_B32, tracked by ASYNCcnt) — no VGPR round trip.
// ---------------------------------------------------------------------------
__global__ __launch_bounds__(384) void patch_embed_kernel(
    const float* __restrict__ x, const float* __restrict__ pw,
    const float* __restrict__ pb, float* __restrict__ h)
{
  __shared__ __align__(16) float patch[768];
  const int bl = blockIdx.x;                // b*196 + l
  const int b = bl / SEQ, l = bl % SEQ;
  const int py = l / 14, px = l % 14;
  const int tid = threadIdx.x;
  for (int idx = tid; idx < 768; idx += 384) {
    const int c = idx >> 8, rem = idx & 255, i = rem >> 4, j = rem & 15;
    const float* gp =
        &x[(((long)b * 3 + c) * 224 + py * 16 + i) * 224 + px * 16 + j];
    const unsigned lds_off = (unsigned)(size_t)&patch[idx];  // low 32b = LDS addr
    asm volatile("global_load_async_to_lds_b32 %0, %1, off"
                 :: "v"(lds_off), "v"(gp) : "memory");
  }
  asm volatile("s_wait_asynccnt 0" ::: "memory");
  __syncthreads();
  const float* wr = pw + (long)tid * 768;   // (c,i,j) contiguous, matches LDS
  float s = pb[tid];
#pragma unroll 8
  for (int k = 0; k < 768; ++k) s += patch[k] * wr[k];
  h[(long)bl * DIM + tid] = s;
}

// ---------------------------------------------------------------------------
// Depthwise causal conv (K=4) over seq + SiLU.  xz is (TOK, 768): x half.
// ---------------------------------------------------------------------------
__global__ void conv_silu_kernel(const float* __restrict__ xz,
                                 const float* __restrict__ cw,
                                 const float* __restrict__ cb,
                                 float* __restrict__ xa)
{
  const int idx = blockIdx.x * blockDim.x + threadIdx.x;
  if (idx >= TOK * DIM) return;
  const int d = idx % DIM, t = idx / DIM;
  const int l = t % SEQ, b = t / SEQ;
  float s = cb[d];
#pragma unroll
  for (int k = 0; k < KCONV; ++k) {
    const int ls = l + k - (KCONV - 1);
    if (ls >= 0) s += xz[(long)(b * SEQ + ls) * (2 * DIM) + d] * cw[d * KCONV + k];
  }
  xa[idx] = silu(s);
}

// ---------------------------------------------------------------------------
// dt = softplus(dbc[:, :R] @ dtw^T + dtb)   (small K=24, plain FMA)
// ---------------------------------------------------------------------------
__global__ void dtproj_kernel(const float* __restrict__ dbc,
                              const float* __restrict__ dtw,
                              const float* __restrict__ dtb,
                              float* __restrict__ dt)
{
  const int idx = blockIdx.x * blockDim.x + threadIdx.x;
  if (idx >= TOK * DIM) return;
  const int m = idx / DIM, d = idx % DIM;
  const float* dr = dbc + (long)m * 56;
  const float* wr = dtw + d * RRANK;
  float s = dtb[d];
#pragma unroll
  for (int r = 0; r < RRANK; ++r) s += dr[r] * wr[r];
  dt[idx] = (s > 20.f) ? s : log1pf(__expf(s));
}

// ---------------------------------------------------------------------------
// Selective scan: one block per batch, one thread per channel d.
// N=16 state lives in registers; B/C reads are wave-uniform.
// ---------------------------------------------------------------------------
__global__ __launch_bounds__(384) void scan_kernel(
    const float* __restrict__ dt, const float* __restrict__ dbc,
    const float* __restrict__ xa, const float* __restrict__ a_log,
    float* __restrict__ ys)
{
  const int b = blockIdx.x, d = threadIdx.x;
  float Av[NST], hst[NST];
#pragma unroll
  for (int n = 0; n < NST; ++n) {
    Av[n] = -__expf(a_log[d * NST + n]);
    hst[n] = 0.f;
  }
  for (int l = 0; l < SEQ; ++l) {
    const long base = (long)(b * SEQ + l);
    const float dtv = dt[base * DIM + d];
    const float xv  = xa[base * DIM + d];
    const float dx  = dtv * xv;
    const float* bc = dbc + base * 56;
    float accv = 0.f;
#pragma unroll
    for (int n = 0; n < NST; ++n) {
      hst[n] = hst[n] * __expf(dtv * Av[n]) + dx * bc[RRANK + n];
      accv  += hst[n] * bc[RRANK + NST + n];
    }
    ys[base * DIM + d] = accv;
  }
}

// ---------------------------------------------------------------------------
// y = (ys + D_skip * xa) * silu(z);   z from second half of xz.
// ---------------------------------------------------------------------------
__global__ void gate_kernel(const float* __restrict__ ys,
                            const float* __restrict__ xa,
                            const float* __restrict__ xz,
                            const float* __restrict__ dsk,
                            float* __restrict__ y)
{
  const int idx = blockIdx.x * blockDim.x + threadIdx.x;
  if (idx >= TOK * DIM) return;
  const int d = idx % DIM;
  const float z = xz[(long)(idx / DIM) * (2 * DIM) + DIM + d];
  y[idx] = (ys[idx] + dsk[d] * xa[idx]) * silu(z);
}

// hid = silu(g1) * g2  (g1 already SiLU'd in GEMM epilogue)
__global__ void mul_kernel(const float* __restrict__ g1,
                           const float* __restrict__ g2,
                           float* __restrict__ o, int n)
{
  const int idx = blockIdx.x * blockDim.x + threadIdx.x;
  if (idx < n) o[idx] = g1[idx] * g2[idx];
}

// ---------------------------------------------------------------------------
extern "C" void kernel_launch(void* const* d_in, const int* in_sizes, int n_in,
                              void* d_out, int out_size, void* d_ws, size_t ws_size,
                              hipStream_t stream)
{
  const float* x        = (const float*)d_in[0];
  const float* patch_w  = (const float*)d_in[1];
  const float* patch_b  = (const float*)d_in[2];
  const float* norm1_w  = (const float*)d_in[3];
  const float* norm1_b  = (const float*)d_in[4];
  const float* in_proj  = (const float*)d_in[5];
  const float* conv_w   = (const float*)d_in[6];
  const float* conv_b   = (const float*)d_in[7];
  const float* x_proj   = (const float*)d_in[8];
  const float* dt_w     = (const float*)d_in[9];
  const float* dt_b     = (const float*)d_in[10];
  const float* A_log    = (const float*)d_in[11];
  const float* D_skip   = (const float*)d_in[12];
  const float* out_proj = (const float*)d_in[13];
  const float* norm2_w  = (const float*)d_in[14];
  const float* norm2_b  = (const float*)d_in[15];
  const float* w1_w     = (const float*)d_in[16];
  const float* w1_b     = (const float*)d_in[17];
  const float* w2_w     = (const float*)d_in[18];
  const float* w2_b     = (const float*)d_in[19];
  const float* w3_w     = (const float*)d_in[20];
  const float* w3_b     = (const float*)d_in[21];
  const float* normf_w  = (const float*)d_in[22];
  const float* normf_b  = (const float*)d_in[23];

  float* ws  = (float*)d_ws;
  float* h   = ws;                 // TOK*DIM
  float* tmp = ws + 602112;        // LN output (shared)
  float* xz  = ws + 1204224;       // TOK*768
  float* xa  = ws + 2408448;       // TOK*DIM
  float* dbc = ws + 3010560;       // TOK*56
  float* dt  = ws + 3098368;       // TOK*DIM
  float* ys  = ws + 3700480;       // TOK*DIM
  float* y   = ws + 4302592;       // TOK*DIM
  float* g1  = ws + 4904704;       // TOK*HID
  float* g2  = ws + 6510336;       // TOK*HID
  float* hid = ws + 8115968;       // TOK*HID   (end: 9721600 floats ~= 39MB)

  const int TOT  = TOK * DIM;           // 602112
  const int TOTH = TOK * HID;           // 1605632
  const dim3 blk256(256);
  const int  gM = (TOK + 127) / 128;    // 13

  patch_embed_kernel<<<TOK, 384, 0, stream>>>(x, patch_w, patch_b, h);

  for (int i = 0; i < DEPTH; ++i) {
    // ---- Mamba block ----
    ln_kernel<<<TOK, 128, 0, stream>>>(h, norm1_w + i * DIM, norm1_b + i * DIM, tmp);
    gemm_wmma_bf16<EPI_STORE><<<dim3(gM, 12), blk256, 0, stream>>>(
        tmp, in_proj + (long)i * 2 * DIM * DIM, nullptr, xz, TOK, 2 * DIM, DIM, 2 * DIM);
    conv_silu_kernel<<<(TOT + 255) / 256, blk256, 0, stream>>>(
        xz, conv_w + (long)i * DIM * KCONV, conv_b + i * DIM, xa);
    gemm_wmma_bf16<EPI_STORE><<<dim3(gM, 1), blk256, 0, stream>>>(
        xa, x_proj + (long)i * 56 * DIM, nullptr, dbc, TOK, 56, DIM, 56);
    dtproj_kernel<<<(TOT + 255) / 256, blk256, 0, stream>>>(
        dbc, dt_w + (long)i * DIM * RRANK, dt_b + i * DIM, dt);
    scan_kernel<<<BATCH, DIM, 0, stream>>>(
        dt, dbc, xa, A_log + (long)i * DIM * NST, ys);
    gate_kernel<<<(TOT + 255) / 256, blk256, 0, stream>>>(ys, xa, xz, D_skip + i * DIM, y);
    gemm_wmma_bf16<EPI_RESADD><<<dim3(gM, 6), blk256, 0, stream>>>(
        y, out_proj + (long)i * DIM * DIM, nullptr, h, TOK, DIM, DIM, DIM);

    // ---- gated MLP ----
    ln_kernel<<<TOK, 128, 0, stream>>>(h, norm2_w + i * DIM, norm2_b + i * DIM, tmp);
    gemm_wmma_bf16<EPI_SILU><<<dim3(gM, 16), blk256, 0, stream>>>(
        tmp, w1_w + (long)i * HID * DIM, w1_b + i * HID, g1, TOK, HID, DIM, HID);
    gemm_wmma_bf16<EPI_STORE><<<dim3(gM, 16), blk256, 0, stream>>>(
        tmp, w2_w + (long)i * HID * DIM, w2_b + i * HID, g2, TOK, HID, DIM, HID);
    mul_kernel<<<(TOTH + 255) / 256, blk256, 0, stream>>>(g1, g2, hid, TOTH);
    gemm_wmma_bf16<EPI_RESADD><<<dim3(gM, 6), blk256, 0, stream>>>(
        hid, w3_w + (long)i * DIM * HID, w3_b + i * DIM, h, TOK, DIM, HID, DIM);
  }

  ln_kernel<<<TOK, 128, 0, stream>>>(h, normf_w, normf_b, (float*)d_out);
}
